// GCNN2_39565238731080
// MI455X (gfx1250) — matrix operand order repaired and checked by reading, not compiled, not run on hardware
//
#include <hip/hip_runtime.h>
#include <hip/hip_bf16.h>
#include <math.h>

#define ALPHA_GCN 0.1f
#define EPS_BN    1e-5f

typedef __attribute__((ext_vector_type(16))) _Float16 v16h;
typedef __attribute__((ext_vector_type(8)))  float    v8f;

// ---------------- utility ----------------
__global__ void fill_f32(float* __restrict__ p, float v, int n) {
  int i = blockIdx.x * blockDim.x + threadIdx.x;
  if (i < n) p[i] = v;
}

// ---------------- edge-weight MLP ----------------
__global__ void edge_lin(const float* __restrict__ in, const float* __restrict__ W,
                         const float* __restrict__ bias, float* __restrict__ hbuf,
                         float* __restrict__ gstats, int E, int din, int dout) {
  __shared__ float s_sum[8], s_ss[8];
  int t = threadIdx.x;
  if (t < 8) { s_sum[t] = 0.f; s_ss[t] = 0.f; }
  __syncthreads();
  int e = blockIdx.x * blockDim.x + t;
  if (e < E) {
    float xin[5];
    for (int k = 0; k < din; ++k) xin[k] = in[(long long)e * din + k];
    for (int d = 0; d < dout; ++d) {
      float acc = bias[d];
      for (int k = 0; k < din; ++k) acc += xin[k] * W[k * dout + d];
      hbuf[(long long)e * dout + d] = acc;
      float rv = fmaxf(acc, 0.f);
      atomicAdd(&s_sum[d], rv);
      atomicAdd(&s_ss[d], rv * rv);
    }
  }
  __syncthreads();
  if (t < dout) {
    atomicAdd(&gstats[t],      s_sum[t]);
    atomicAdd(&gstats[64 + t], s_ss[t]);
  }
}

// bn(relu(h)) + h  (in place)
__global__ void edge_bn(float* __restrict__ hbuf, const float* __restrict__ gstats,
                        const float* __restrict__ g, const float* __restrict__ bb,
                        int total, int dout, float invCount) {
  int idx = blockIdx.x * blockDim.x + threadIdx.x;
  if (idx < total) {
    int d = idx % dout;
    float mu  = gstats[d] * invCount;
    float var = gstats[64 + d] * invCount - mu * mu;
    float v  = hbuf[idx];
    float rv = fmaxf(v, 0.f);
    hbuf[idx] = (rv - mu) * rsqrtf(var + EPS_BN) * g[d] + bb[d] + v;
  }
}

__global__ void edge_sigmoid(const float* __restrict__ in, float* __restrict__ ew, int E) {
  int e = blockIdx.x * blockDim.x + threadIdx.x;
  if (e < E) ew[e] = 1.f / (1.f + expf(-in[e]));
}

// ---------------- gcn_norm ----------------
__global__ void deg_accum(const float* __restrict__ ew, const int* __restrict__ col,
                          float* __restrict__ deg, int E) {
  int e = blockIdx.x * blockDim.x + threadIdx.x;
  if (e < E) atomicAdd(&deg[col[e]], ew[e]);
}

__global__ void deg_rsqrt(const float* __restrict__ deg, float* __restrict__ dis, int N) {
  int i = blockIdx.x * blockDim.x + threadIdx.x;
  if (i < N) dis[i] = rsqrtf(fmaxf(deg[i], 1e-12f));
}

// ---------------- first linear: h0 = x @ W_first + b ----------------
__global__ void first_lin(const float* __restrict__ x, const float* __restrict__ W,
                          const float* __restrict__ b, float* __restrict__ h0, int N) {
  int idx = blockIdx.x * blockDim.x + threadIdx.x;
  if (idx < N * 64) {
    int n = idx >> 6, f = idx & 63;
    h0[idx] = x[n * 2 + 0] * W[f] + x[n * 2 + 1] * W[64 + f] + b[f];
  }
}

// ---------------- GCN aggregate: agg[col] += h[row]*norm (incl. self loops) --------
// 32 lanes/edge, float2 per lane; sequential streams prefetched ahead of atomics.
__global__ void gcn_scatter(const float* __restrict__ h, const float* __restrict__ dis,
                            const float* __restrict__ ew, const int* __restrict__ row,
                            const int* __restrict__ col, float* __restrict__ agg,
                            int E, int N) {
  int gid  = blockIdx.x * blockDim.x + threadIdx.x;
  int item = gid >> 5;
  int f    = (gid & 31) * 2;
  if (item < E) {
    if (item + 16384 < E) {
      __builtin_prefetch(&ew[item + 16384], 0, 0);
      __builtin_prefetch(&row[item + 16384], 0, 0);
      __builtin_prefetch(&col[item + 16384], 0, 0);
    }
    int r = row[item], c = col[item];
    float nrm = dis[r] * ew[item] * dis[c];
    float2 hv = *(const float2*)(h + (long long)r * 64 + f);
    atomicAdd(&agg[(long long)c * 64 + f],     hv.x * nrm);
    atomicAdd(&agg[(long long)c * 64 + f + 1], hv.y * nrm);
  } else if (item < E + N) {
    int i = item - E;
    float d = dis[i];
    float nrm = d * d;
    float2 hv = *(const float2*)(h + (long long)i * 64 + f);
    atomicAdd(&agg[(long long)i * 64 + f],     hv.x * nrm);
    atomicAdd(&agg[(long long)i * 64 + f + 1], hv.y * nrm);
  }
}

// ---------------- WMMA GEMM: y = relu(((1-a)agg + a h0) @ W), fused relu-stats ------
// Block = 32 rows x 64 cols (8 waves = 2 M-tiles x 4 N-tiles). A and B staged in LDS
// in WMMA fragment order (f16), fetched as contiguous v16h per lane.
__global__ __launch_bounds__(256) void gcn_gemm(
    const float* __restrict__ agg, const float* __restrict__ h0,
    const float* __restrict__ W, float* __restrict__ y,
    float* __restrict__ gstats, int nrows) {
  // [tile][kk][lane][half]: per-lane fragment contiguous (32B)
  __shared__ __align__(32) _Float16 alds[2][2][32][16];
  __shared__ __align__(32) _Float16 blds[4][2][32][16];
  int t = threadIdx.x;
  int rowBase = blockIdx.x * 32;

  // Stage B = W (64x64) -> fragment order: lane = N + 16*((K>>3)&1), elem = (K>>4)*8 + (K&7)
  for (int i = t; i < 64 * 64; i += 256) {
    int kglob = i >> 6;                 // 0..63
    int ncol  = i & 63;                 // 0..63
    blds[ncol >> 4][kglob >> 5][((kglob >> 3) & 1) * 16 + (ncol & 15)][((kglob & 31) >> 4) * 8 + (kglob & 7)] = (_Float16)W[i];
  }
  // Stage A = (1-a)*agg + a*h0 (32 rows) -> fragment order per ISA 16-bit A layout
  for (int i = t; i < 32 * 64; i += 256) {
    int ml    = i >> 6;                 // local row 0..31
    int kglob = i & 63;
    long long src = (long long)(rowBase + ml) * 64 + kglob;
    float av = (1.0f - ALPHA_GCN) * agg[src] + ALPHA_GCN * h0[src];
    int kk = kglob >> 5, Kl = kglob & 31;
    int g  = (Kl >> 3) & 1;
    int kr = Kl - g * 8;                // in {0..7} U {16..23}
    int v  = (kr >= 16) ? (4 + ((kr - 16) >> 1)) : (kr >> 1);
    int p  = kr & 1;
    alds[ml >> 4][kk][g * 16 + (ml & 15)][2 * v + p] = (_Float16)av;
  }
  __syncthreads();

  int wave = t >> 5, lane = t & 31;
  int mtile = wave >> 2, ntile = wave & 3;
  v8f c = {};
#pragma unroll
  for (int kk = 0; kk < 2; ++kk) {
    v16h a = *(const v16h*)(&alds[mtile][kk][lane][0]);
    v16h b = *(const v16h*)(&blds[ntile][kk][lane][0]);
    c = __builtin_amdgcn_wmma_f32_16x16x32_f16(false, a, false, b, (short)0, c,
                                               false, false);
  }

  // C layout: lane%16 = N column, VGPR r = M row (lanes>=16 -> M+8)
  int m0   = rowBase + mtile * 16;
  int feat = ntile * 16 + (lane & 15);
  int gsel = lane >> 4;
  float sum = 0.f, ss = 0.f;
#pragma unroll
  for (int r = 0; r < 8; ++r) {
    int mm = gsel * 8 + r;
    float v = fmaxf(c[r], 0.f);
    y[(long long)(m0 + mm) * 64 + feat] = v;
    sum += v; ss += v * v;
  }
  atomicAdd(&gstats[feat],      sum);
  atomicAdd(&gstats[64 + feat], ss);
}

// h = bn(y): y already relu'ed, stats over relu(y); float4 per thread (b128 traffic)
__global__ void gcn_bn(const float4* __restrict__ y4, const float* __restrict__ gstats,
                       const float* __restrict__ g, const float* __restrict__ bb,
                       float4* __restrict__ h4, int total4, float invN) {
  int idx = blockIdx.x * blockDim.x + threadIdx.x;
  if (idx < total4) {
    int fb = (idx & 15) * 4;
    float4 v = y4[idx];
    float4 o;
    {
      float mu = gstats[fb] * invN, var = gstats[64 + fb] * invN - mu * mu;
      o.x = (v.x - mu) * rsqrtf(var + EPS_BN) * g[fb] + bb[fb];
    }
    {
      float mu = gstats[fb + 1] * invN, var = gstats[64 + fb + 1] * invN - mu * mu;
      o.y = (v.y - mu) * rsqrtf(var + EPS_BN) * g[fb + 1] + bb[fb + 1];
    }
    {
      float mu = gstats[fb + 2] * invN, var = gstats[64 + fb + 2] * invN - mu * mu;
      o.z = (v.z - mu) * rsqrtf(var + EPS_BN) * g[fb + 2] + bb[fb + 2];
    }
    {
      float mu = gstats[fb + 3] * invN, var = gstats[64 + fb + 3] * invN - mu * mu;
      o.w = (v.w - mu) * rsqrtf(var + EPS_BN) * g[fb + 3] + bb[fb + 3];
    }
    h4[idx] = o;
  }
}

// ---------------- pooling ----------------
__global__ void pool_accum(const float* __restrict__ h, const int* __restrict__ batch,
                           float* __restrict__ xa, int N) {
  int idx = blockIdx.x * blockDim.x + threadIdx.x;
  if (idx < N * 64) {
    int n = idx >> 6, f = idx & 63;
    atomicAdd(&xa[batch[n] * 64 + f], h[idx]);
  }
}

__global__ void pool_sigmoid(float* __restrict__ xa, float* __restrict__ out2, int total) {
  int idx = blockIdx.x * blockDim.x + threadIdx.x;
  if (idx < total) {
    float v = 1.f / (1.f + expf(-xa[idx]));
    xa[idx]   = v;
    out2[idx] = v;   // second output: x_aggr [128,64]
  }
}

// ---------------- head MLP: single workgroup, G=128 rows ----------------
__global__ __launch_bounds__(256) void head_mlp(
    const float* __restrict__ xa, const float* __restrict__ x10,
    const float* W0, const float* W1, const float* W2,
    const float* b0, const float* b1, const float* b2,
    const float* g0, const float* g1, const float* g2,
    const float* bb0, const float* bb1, const float* bb2,
    float* __restrict__ zb0, float* __restrict__ zb1, float* __restrict__ out) {
  __shared__ float s_sum[128], s_ss[128];
  int t = threadIdx.x;

  for (int idx = t; idx < 128 * 74; idx += 256) {
    int r = idx / 74, cc = idx % 74;
    zb0[r * 128 + cc] = (cc < 64) ? xa[r * 64 + cc] : x10[r * 10 + (cc - 64)];
  }
  __threadfence(); __syncthreads();

  const int   dins[3]  = {74, 128, 66};
  const int   douts[3] = {128, 66, 4};
  const float* Ws[3]  = {W0, W1, W2};
  const float* bs[3]  = {b0, b1, b2};
  const float* gs[3]  = {g0, g1, g2};
  const float* bbs[3] = {bb0, bb1, bb2};

  float* zin = zb0; float* zout = zb1;
  for (int s = 0; s < 3; ++s) {
    int din = dins[s], dout = douts[s];
    for (int i = t; i < 128; i += 256) { s_sum[i] = 0.f; s_ss[i] = 0.f; }
    __syncthreads();
    for (int idx = t; idx < 128 * dout; idx += 256) {
      int r = idx / dout, cc = idx % dout;
      float acc = bs[s][cc];
      for (int k = 0; k < din; ++k) acc += zin[r * 128 + k] * Ws[s][k * dout + cc];
      zout[r * 128 + cc] = acc;
      float rv = fmaxf(acc, 0.f);
      atomicAdd(&s_sum[cc], rv);
      atomicAdd(&s_ss[cc], rv * rv);
    }
    __threadfence(); __syncthreads();
    for (int idx = t; idx < 128 * dout; idx += 256) {
      int r = idx / dout, cc = idx % dout;
      float v   = zout[r * 128 + cc];
      float mu  = s_sum[cc] * (1.f / 128.f);
      float var = s_ss[cc] * (1.f / 128.f) - mu * mu;
      float rv  = fmaxf(v, 0.f);
      zout[r * 128 + cc] = (rv - mu) * rsqrtf(var + EPS_BN) * gs[s][cc] + bbs[s][cc] + v;
    }
    __threadfence(); __syncthreads();
    float* tmp = zin; zin = zout; zout = tmp;
  }
  for (int idx = t; idx < 128 * 4; idx += 256) {
    int r = idx / 4, cc = idx % 4;
    out[idx] = 1.f / (1.f + expf(-zin[r * 128 + cc]));
  }
}

// ---------------- launch ----------------
extern "C" void kernel_launch(void* const* d_in, const int* in_sizes, int n_in,
                              void* d_out, int out_size, void* d_ws, size_t ws_size,
                              hipStream_t stream) {
  (void)in_sizes; (void)n_in; (void)out_size; (void)ws_size;
  const int N = 100000, E = 1000000, L = 4, G = 128;

  const float* x         = (const float*)d_in[0];
  const float* edge_attr = (const float*)d_in[1];
  const float* x10       = (const float*)d_in[2];
  const float* W_first   = (const float*)d_in[3];
  const float* b_first   = (const float*)d_in[4];
  const float* gcn_W     = (const float*)d_in[5];
  const float* gcn_g     = (const float*)d_in[6];
  const float* gcn_b     = (const float*)d_in[7];
  const float* ew_Ws[5], *ew_bs[5], *ew_gs[5], *ew_bbs[5];
  for (int i = 0; i < 5; ++i) {
    ew_Ws[i]  = (const float*)d_in[8 + i];
    ew_bs[i]  = (const float*)d_in[13 + i];
    ew_gs[i]  = (const float*)d_in[18 + i];
    ew_bbs[i] = (const float*)d_in[23 + i];
  }
  const float* mlp_Ws[3], *mlp_bs[3], *mlp_gs[3], *mlp_bbs[3];
  for (int i = 0; i < 3; ++i) {
    mlp_Ws[i]  = (const float*)d_in[28 + i];
    mlp_bs[i]  = (const float*)d_in[31 + i];
    mlp_gs[i]  = (const float*)d_in[34 + i];
    mlp_bbs[i] = (const float*)d_in[37 + i];
  }
  const int* edge_index = (const int*)d_in[40];
  const int* batch      = (const int*)d_in[41];
  const int* row = edge_index;
  const int* col = edge_index + E;

  float* ws = (float*)d_ws;
  float* ebuf0 = ws;  ws += (size_t)E * 5;
  float* ebuf1 = ws;  ws += (size_t)E * 5;
  float* ew    = ws;  ws += E;
  float* deg   = ws;  ws += N;
  float* dis   = ws;  ws += N;
  float* h0    = ws;  ws += (size_t)N * 64;
  float* h     = ws;  ws += (size_t)N * 64;
  float* agg   = ws;  ws += (size_t)N * 64;
  float* ybuf  = ws;  ws += (size_t)N * 64;
  float* stats = ws;  ws += 256;
  float* xa    = ws;  ws += (size_t)G * 64;
  float* zb0   = ws;  ws += (size_t)G * 128;
  float* zb1   = ws;  ws += (size_t)G * 128;

  auto nb = [](long long n) { return (int)((n + 255) / 256); };

  // ---- edge-weight MLP (5 BN-separated stages, ping-pong buffers) ----
  const int edims[6] = {5, 4, 3, 2, 1, 1};
  const float* cur = edge_attr;
  float* bufs[2] = {ebuf0, ebuf1};
  for (int i = 0; i < 5; ++i) {
    int din = edims[i], dout = edims[i + 1];
    float* hb = bufs[i & 1];
    fill_f32<<<1, 256, 0, stream>>>(stats, 0.f, 128);
    edge_lin<<<nb(E), 256, 0, stream>>>(cur, ew_Ws[i], ew_bs[i], hb, stats, E, din, dout);
    edge_bn<<<nb((long long)E * dout), 256, 0, stream>>>(hb, stats, ew_gs[i], ew_bbs[i],
                                                         E * dout, dout, 1.f / (float)E);
    cur = hb;
  }
  edge_sigmoid<<<nb(E), 256, 0, stream>>>(cur, ew, E);

  // ---- gcn_norm (deg init 1.0 covers the self-loop weight) ----
  fill_f32<<<nb(N), 256, 0, stream>>>(deg, 1.f, N);
  deg_accum<<<nb(E), 256, 0, stream>>>(ew, col, deg, E);
  deg_rsqrt<<<nb(N), 256, 0, stream>>>(deg, dis, N);

  // ---- h0 ----
  first_lin<<<nb((long long)N * 64), 256, 0, stream>>>(x, W_first, b_first, h0, N);

  // ---- 4 GCN2Conv layers ----
  const float* hin = h0;
  int gemmBlocks = N / 32;                       // 32 rows per block
  for (int l = 0; l < L; ++l) {
    fill_f32<<<nb((long long)N * 64), 256, 0, stream>>>(agg, 0.f, N * 64);
    gcn_scatter<<<nb((long long)(E + N) * 32), 256, 0, stream>>>(hin, dis, ew, row, col,
                                                                 agg, E, N);
    fill_f32<<<1, 256, 0, stream>>>(stats, 0.f, 128);
    gcn_gemm<<<gemmBlocks, 256, 0, stream>>>(agg, h0, gcn_W + (size_t)l * 64 * 64,
                                             ybuf, stats, N);
    gcn_bn<<<nb((long long)N * 16), 256, 0, stream>>>((const float4*)ybuf, stats,
                                                      gcn_g + l * 64, gcn_b + l * 64,
                                                      (float4*)h, N * 16,
                                                      1.f / (float)N);
    hin = h;
  }

  // ---- pooling + head ----
  float* out = (float*)d_out;
  fill_f32<<<nb(G * 64), 256, 0, stream>>>(xa, 0.f, G * 64);
  pool_accum<<<nb((long long)N * 64), 256, 0, stream>>>(h, batch, xa, N);
  pool_sigmoid<<<nb(G * 64), 256, 0, stream>>>(xa, out + G * 4, G * 64);
  head_mlp<<<1, 256, 0, stream>>>(xa, x10, mlp_Ws[0], mlp_Ws[1], mlp_Ws[2],
                                  mlp_bs[0], mlp_bs[1], mlp_bs[2],
                                  mlp_gs[0], mlp_gs[1], mlp_gs[2],
                                  mlp_bbs[0], mlp_bbs[1], mlp_bbs[2],
                                  zb0, zb1, out);
}
